// ObjectDetection_4131758538965
// MI455X (gfx1250) — matrix operand order repaired and checked by reading, not compiled, not run on hardware
//
#include <hip/hip_runtime.h>
#include <math.h>

// ---------------- problem constants ----------------
#define C        256
#define NTOK     5376          // tokens per batch (64*64 + 32*32 + 16*16)
#define BATCH    8
#define MTOT     (BATCH * NTOK)   // 43008, multiple of 16
#define KINST    100
#define NHEAD    (BATCH * KINST)  // 800, multiple of 16
#define NCLS     80

typedef __attribute__((ext_vector_type(16))) __bf16 v16bf;
typedef __attribute__((ext_vector_type(8)))  float  v8f;

union FragB16 { v16bf v; uint4 q[2]; unsigned int d[8]; unsigned short u[16]; };
union FragF   { v8f   v; float f[8]; };

__device__ __forceinline__ unsigned short f2bf(float x) {
    __bf16 h = (__bf16)x;                       // native cvt, RNE
    return __builtin_bit_cast(unsigned short, h);
}
__device__ __forceinline__ float bf2f(unsigned short u) {
    unsigned int x = ((unsigned int)u) << 16;
    return __builtin_bit_cast(float, x);
}

// ---------------- weight conversion ----------------
__global__ void k_cvt(const float* __restrict__ in, unsigned short* __restrict__ out, int n) {
    int i = blockIdx.x * blockDim.x + threadIdx.x;
    if (i < n) out[i] = f2bf(in[i]);
}
// in [K][N] f32 -> out [N][K] bf16
__global__ void k_cvt_t(const float* __restrict__ in, unsigned short* __restrict__ out, int K, int N) {
    int i = blockIdx.x * blockDim.x + threadIdx.x;
    if (i < K * N) { int k = i / N, n = i % N; out[(size_t)n * K + k] = f2bf(in[i]); }
}

// ---------------- LDS-tiled transpose+convert: [B][CIN][HW] f32 -> [B][HW][CIN] bf16 ----
template<int CIN, int HW>
__global__ void k_transpose(const float* __restrict__ x, unsigned short* __restrict__ xT)
{
    __shared__ unsigned short tile[32][33];     // +1 pad: no bank conflicts
    int b  = blockIdx.z;
    int c0 = blockIdx.x * 32;
    int h0 = blockIdx.y * 32;
    const float* src = x + (size_t)b * CIN * HW;
    unsigned short* dst = xT + (size_t)b * HW * CIN;
    int tx = threadIdx.x;                       // 0..31
    int ty = threadIdx.y;                       // 0..7
#pragma unroll
    for (int i = ty; i < 32; i += 8)
        tile[i][tx] = f2bf(src[(size_t)(c0 + i) * HW + h0 + tx]);   // coalesced read along hw
    __syncthreads();
#pragma unroll
    for (int i = ty; i < 32; i += 8)
        dst[(size_t)(h0 + i) * CIN + c0 + tx] = tile[tx][i];        // coalesced write along c
}

// ---- pipelined fragment loaders (ISA 7.12.2 layouts; contiguous b128 runs) ----
__device__ __forceinline__ void loadA(FragB16& a, const uint4* xrow, int k0, int hh) {
    int qi = k0 >> 3;
    a.q[0] = xrow[qi + hh];
    a.q[1] = xrow[qi + hh + 2];
}
__device__ __forceinline__ void loadB(FragB16& b, const uint4* wrow, int k0, int hh) {
    int qi = k0 >> 3;
    b.q[0] = wrow[qi + hh];
    b.q[1] = wrow[qi + hh + 2];
}

// ---------------- lateral 1x1 conv + BN -> bf16 flat, via WMMA ----------------
// xT: [B*HW][CIN] bf16 (token-major); wt: [256][CIN] bf16 ; flat: [MTOT][256] bf16
// one wave per block; wave computes 16 tokens x 64 out-channels; K-loop software-pipelined
template<int CIN, int HW>
__global__ void k_lateral(const unsigned short* __restrict__ xT,
                          const unsigned short* __restrict__ wt,
                          const float* __restrict__ g, const float* __restrict__ bta,
                          const float* __restrict__ mn, const float* __restrict__ vr,
                          unsigned short* __restrict__ flat, int lvlOff)
{
    constexpr int TN = 4;
    int lane = threadIdx.x;                 // wave32 block
    int tileM = blockIdx.x;                 // 16-token tile (within level, batch-contiguous)
    int tn0  = blockIdx.y * TN;             // first out-channel tile
    int mloc = lane & 15;
    int hh   = lane >> 4;

    const uint4* xrow = (const uint4*)(xT + (size_t)(tileM * 16 + mloc) * CIN);
    const uint4* wrow[TN];
#pragma unroll
    for (int t = 0; t < TN; ++t)
        wrow[t] = (const uint4*)(wt + (size_t)((tn0 + t) * 16 + mloc) * CIN);

    FragF acc[TN];
#pragma unroll
    for (int t = 0; t < TN; ++t)
#pragma unroll
        for (int j = 0; j < 8; ++j) acc[t].f[j] = 0.0f;

    FragB16 a, bm[TN];
    loadA(a, xrow, 0, hh);
#pragma unroll
    for (int t = 0; t < TN; ++t) loadB(bm[t], wrow[t], 0, hh);

#pragma unroll 8
    for (int k0 = 0; k0 < CIN; k0 += 32) {
        FragB16 an, bn[TN];
        if (k0 + 32 < CIN) {                 // issue next step's loads before this step's math
            loadA(an, xrow, k0 + 32, hh);
#pragma unroll
            for (int t = 0; t < TN; ++t) loadB(bn[t], wrow[t], k0 + 32, hh);
        }
#pragma unroll
        for (int t = 0; t < TN; ++t)
            acc[t].v = __builtin_amdgcn_wmma_f32_16x16x32_bf16(false, a.v, false, bm[t].v,
                                                               (short)0, acc[t].v, false, false);
        if (k0 + 32 < CIN) {
            a = an;
#pragma unroll
            for (int t = 0; t < TN; ++t) bm[t] = bn[t];
        }
    }
    // BN (eval) epilogue, write bf16
    int tilesPerBatch = HW >> 4;
    int b   = tileM / tilesPerBatch;
    int hw0 = (tileM % tilesPerBatch) << 4;
    int rowBase = b * NTOK + lvlOff + hw0 + hh * 8;
#pragma unroll
    for (int t = 0; t < TN; ++t) {
        int col = (tn0 + t) * 16 + mloc;
        float sc = g[col] * __frsqrt_rn(vr[col] + 1e-5f);
        float mm = mn[col], bb = bta[col];
#pragma unroll
        for (int r = 0; r < 8; ++r) {
            float y = (acc[t].f[r] - mm) * sc + bb;
            flat[(size_t)(rowBase + r) * C + col] = f2bf(y);
        }
    }
}

// ---------------- generic bf16 WMMA linear: Y f32 = X @ Wt^T + bias ----------------
// X: [M][KD] bf16 ; Wt: [N][KD] bf16 ; Y: [M][N] f32
// one wave per block; wave computes 16 rows x (TN*16) cols; K-loop software-pipelined
template<int TN, int KD>
__global__ void k_linear(const unsigned short* __restrict__ X,
                         const unsigned short* __restrict__ Wt,
                         const float* __restrict__ bias,
                         float* __restrict__ Y, int M, int N)
{
    int lane = threadIdx.x;
    int tm = blockIdx.x;
    int tn0 = blockIdx.y * TN;
    int mloc = lane & 15, hh = lane >> 4;
    const uint4* xrow = (const uint4*)(X + (size_t)(tm * 16 + mloc) * KD);
    const uint4* wrow[TN];
#pragma unroll
    for (int t = 0; t < TN; ++t)
        wrow[t] = (const uint4*)(Wt + (size_t)((tn0 + t) * 16 + mloc) * KD);

    FragF acc[TN];
#pragma unroll
    for (int t = 0; t < TN; ++t)
#pragma unroll
        for (int j = 0; j < 8; ++j) acc[t].f[j] = 0.0f;

    FragB16 a, bm[TN];
    loadA(a, xrow, 0, hh);
#pragma unroll
    for (int t = 0; t < TN; ++t) loadB(bm[t], wrow[t], 0, hh);

#pragma unroll
    for (int k0 = 0; k0 < KD; k0 += 32) {
        FragB16 an, bn[TN];
        if (k0 + 32 < KD) {                  // issue next step's loads before this step's math
            loadA(an, xrow, k0 + 32, hh);
#pragma unroll
            for (int t = 0; t < TN; ++t) loadB(bn[t], wrow[t], k0 + 32, hh);
        }
#pragma unroll
        for (int t = 0; t < TN; ++t)
            acc[t].v = __builtin_amdgcn_wmma_f32_16x16x32_bf16(false, a.v, false, bm[t].v,
                                                               (short)0, acc[t].v, false, false);
        if (k0 + 32 < KD) {
            a = an;
#pragma unroll
            for (int t = 0; t < TN; ++t) bm[t] = bn[t];
        }
    }
    int row0 = tm * 16 + hh * 8;
#pragma unroll
    for (int t = 0; t < TN; ++t) {
        int col = (tn0 + t) * 16 + mloc;
        float bv = bias ? bias[col] : 0.0f;
#pragma unroll
        for (int r = 0; r < 8; ++r)
            Y[(size_t)(row0 + r) * N + col] = acc[t].f[r] + bv;
    }
}

// ---------------- LayerNorm + SiLU, f32 in -> bf16 out (one wave per token) ----------
__global__ void k_ln_silu(const float* __restrict__ Yin,
                          const float* __restrict__ g, const float* __restrict__ b,
                          unsigned short* __restrict__ out, int M)
{
    int wave = (int)((blockIdx.x * blockDim.x + threadIdx.x) >> 5);
    int lane = threadIdx.x & 31;
    if (wave >= M) return;
    const float* row = Yin + (size_t)wave * C;
    float v[8]; float s = 0.f;
#pragma unroll
    for (int j = 0; j < 8; ++j) { v[j] = row[lane + 32 * j]; s += v[j]; }
#pragma unroll
    for (int m = 16; m; m >>= 1) s += __shfl_xor(s, m, 32);
    float mean = s * (1.0f / C);
    float q = 0.f;
#pragma unroll
    for (int j = 0; j < 8; ++j) { float d = v[j] - mean; q += d * d; }
#pragma unroll
    for (int m = 16; m; m >>= 1) q += __shfl_xor(q, m, 32);
    float rstd = __frsqrt_rn(q * (1.0f / C) + 1e-5f);
    unsigned short* orow = out + (size_t)wave * C;
#pragma unroll
    for (int j = 0; j < 8; ++j) {
        int c = lane + 32 * j;
        float y = (v[j] - mean) * rstd * g[c] + b[c];
        orow[c] = f2bf(y / (1.f + __expf(-y)));     // SiLU
    }
}

// ---------------- loc final 256 -> 1 (one wave per token) ----------------
__global__ void k_dot1(const unsigned short* __restrict__ X, const unsigned short* __restrict__ w,
                       const float* __restrict__ bf, float* __restrict__ out, int M)
{
    int wave = (int)((blockIdx.x * blockDim.x + threadIdx.x) >> 5);
    int lane = threadIdx.x & 31;
    if (wave >= M) return;
    const unsigned short* row = X + (size_t)wave * C;
    float s = 0.f;
#pragma unroll
    for (int j = 0; j < 8; ++j) { int c = lane + 32 * j; s += bf2f(row[c]) * bf2f(w[c]); }
#pragma unroll
    for (int m = 16; m; m >>= 1) s += __shfl_xor(s, m, 32);
    if (lane == 0) out[wave] = s + bf[0];
}

// ---------------- per-batch iterative top-100 in LDS ----------------
__global__ void k_topk(const float* __restrict__ logits, float* __restrict__ vals, int* __restrict__ idx)
{
    __shared__ float sv[NTOK];
    __shared__ float rv[256];
    __shared__ int   ri[256];
    int b = blockIdx.x, t = threadIdx.x;
    for (int i = t; i < NTOK; i += 256) sv[i] = logits[b * NTOK + i];
    __syncthreads();
    for (int k = 0; k < KINST; ++k) {
        float bv = -3.402823466e38f; int bi = 0x7fffffff;
        for (int i = t; i < NTOK; i += 256) {
            float v = sv[i];
            if (v > bv || (v == bv && i < bi)) { bv = v; bi = i; }
        }
        rv[t] = bv; ri[t] = bi;
        __syncthreads();
        for (int s = 128; s; s >>= 1) {
            if (t < s && (rv[t + s] > rv[t] || (rv[t + s] == rv[t] && ri[t + s] < ri[t]))) {
                rv[t] = rv[t + s]; ri[t] = ri[t + s];
            }
            __syncthreads();
        }
        if (t == 0) { vals[b * KINST + k] = rv[0]; idx[b * KINST + k] = ri[0]; sv[ri[0]] = -3.402823466e38f; }
        __syncthreads();
    }
}

// ---------------- gather selected tokens (bf16 rows) ----------------
__global__ void k_gather(const unsigned short* __restrict__ flat, const int* __restrict__ idx,
                         unsigned short* __restrict__ g)
{
    int r = blockIdx.x;                    // 0..799
    int b = r / KINST;
    int n = idx[r];
    g[(size_t)r * C + threadIdx.x] = flat[((size_t)b * NTOK + n) * C + threadIdx.x];
}

// ---------------- argmax over 80 classes (one wave per token) ----------------
__global__ void k_argmax80(const float* __restrict__ logits, float* __restrict__ cls, int M)
{
    int wave = (int)((blockIdx.x * blockDim.x + threadIdx.x) >> 5);
    int lane = threadIdx.x & 31;
    if (wave >= M) return;
    const float* row = logits + (size_t)wave * NCLS;
    float bv = -3.402823466e38f; int bi = 0x7fffffff;
    for (int c = lane; c < NCLS; c += 32) {
        float v = row[c];
        if (v > bv || (v == bv && c < bi)) { bv = v; bi = c; }
    }
#pragma unroll
    for (int m = 16; m; m >>= 1) {
        float ov = __shfl_xor(bv, m, 32); int oi = __shfl_xor(bi, m, 32);
        if (ov > bv || (ov == bv && oi < bi)) { bv = ov; bi = oi; }
    }
    if (lane == 0) cls[wave] = (float)bi;
}

// ---------------- box final 256 -> 4 (one wave per token, 8 lanes per output) --------
__global__ void k_dot4(const unsigned short* __restrict__ X, const unsigned short* __restrict__ wt,
                       const float* __restrict__ bf4, float* __restrict__ out, int M)
{
    int wave = (int)((blockIdx.x * blockDim.x + threadIdx.x) >> 5);
    int lane = threadIdx.x & 31;
    if (wave >= M) return;
    int o = lane >> 3, j = lane & 7;
    const unsigned short* row = X + (size_t)wave * C;
    const unsigned short* w = wt + o * C;
    float s = 0.f;
    for (int c = j; c < C; c += 8) s += bf2f(row[c]) * bf2f(w[c]);
#pragma unroll
    for (int m = 4; m; m >>= 1) s += __shfl_xor(s, m, 8);
    if (j == 0) out[(size_t)wave * 4 + o] = s + bf4[o];
}

// ---------------- finalize: sigmoid, counts, box decode ----------------
__global__ void k_finalize(const float* __restrict__ vals, const int* __restrict__ idx,
                           const float* __restrict__ clsArg, const float* __restrict__ boxRaw,
                           const int* __restrict__ fh, const int* __restrict__ fw,
                           float* __restrict__ out)
{
    __shared__ int cnt;
    int b = blockIdx.x, k = threadIdx.x;   // blockDim 128
    if (k == 0) cnt = 0;
    __syncthreads();
    if (k < KINST) {
        int r = b * KINST + k;
        float score = 1.f / (1.f + __expf(-vals[r]));
        out[8 + r] = score;
        if (score > 0.5f) atomicAdd(&cnt, 1);
        out[808 + r] = clsArg[r];
        int n = idx[r];
        int base, dim;
        if (n < 4096)      { base = 0;    dim = 64; }
        else if (n < 5120) { base = 4096; dim = 32; }
        else               { base = 5120; dim = 16; }
        int loc = n - base;
        int row = loc / dim, col = loc % dim;
        float inv = 1.0f / (float)dim;
        float gx = (col + 0.5f) * inv, gy = (row + 0.5f) * inv;
        float off[4] = { gx, gy, gx, gy };
        float scl[4] = { -0.5f * inv, -0.5f * inv, 0.5f * inv, 0.5f * inv };
        float full[4] = { (float)fw[0], (float)fh[0], (float)fw[0], (float)fh[0] };
#pragma unroll
        for (int j = 0; j < 4; ++j)
            out[1608 + r * 4 + j] = (off[j] + scl[j] * __expf(boxRaw[r * 4 + j])) * full[j];
    }
    __syncthreads();
    if (k == 0) out[b] = (float)cnt;
}

// =================================================================================
extern "C" void kernel_launch(void* const* d_in, const int* in_sizes, int n_in,
                              void* d_out, int out_size, void* d_ws, size_t ws_size,
                              hipStream_t stream)
{
    const float* x3     = (const float*)d_in[0];
    const float* x4     = (const float*)d_in[1];
    const float* x5     = (const float*)d_in[2];
    const float* lat_w3 = (const float*)d_in[3];
    const float* lat_w4 = (const float*)d_in[4];
    const float* lat_w5 = (const float*)d_in[5];
    const float* bn_g   = (const float*)d_in[6];
    const float* bn_b   = (const float*)d_in[7];
    const float* bn_m   = (const float*)d_in[8];
    const float* bn_v   = (const float*)d_in[9];
    const float* loc_Ws = (const float*)d_in[10];
    const float* loc_bs = (const float*)d_in[11];
    const float* loc_lng= (const float*)d_in[12];
    const float* loc_lnb= (const float*)d_in[13];
    const float* loc_Wf = (const float*)d_in[14];
    const float* loc_bf = (const float*)d_in[15];
    const float* cls_Ws = (const float*)d_in[16];
    const float* cls_bs = (const float*)d_in[17];
    const float* cls_lng= (const float*)d_in[18];
    const float* cls_lnb= (const float*)d_in[19];
    const float* cls_Wf = (const float*)d_in[20];
    const float* cls_bf = (const float*)d_in[21];
    const float* box_Ws = (const float*)d_in[22];
    const float* box_bs = (const float*)d_in[23];
    const float* box_lng= (const float*)d_in[24];
    const float* box_lnb= (const float*)d_in[25];
    const float* box_Wf = (const float*)d_in[26];
    const float* box_bf = (const float*)d_in[27];
    const int*   full_h = (const int*)d_in[28];
    const int*   full_w = (const int*)d_in[29];
    float* out = (float*)d_out;

    char* ws = (char*)d_ws;
    size_t off = 0;
    auto alloc = [&](size_t bytes) -> void* {
        void* p = ws + off;
        off += (bytes + 255) & ~(size_t)255;
        return p;
    };
    unsigned short* latW3b = (unsigned short*)alloc(256 * 256  * 2);
    unsigned short* latW4b = (unsigned short*)alloc(256 * 512  * 2);
    unsigned short* latW5b = (unsigned short*)alloc(256 * 1024 * 2);
    unsigned short* locWt  = (unsigned short*)alloc(4 * 65536 * 2);
    unsigned short* locWfT = (unsigned short*)alloc(256 * 2);
    unsigned short* clsWt  = (unsigned short*)alloc(4 * 65536 * 2);
    unsigned short* clsWfT = (unsigned short*)alloc(NCLS * 256 * 2);
    unsigned short* boxWt  = (unsigned short*)alloc(4 * 65536 * 2);
    unsigned short* boxWfT = (unsigned short*)alloc(4 * 256 * 2);
    unsigned short* xT     = (unsigned short*)alloc((size_t)BATCH * 4096 * 256 * 2); // reused per level
    unsigned short* flatBf = (unsigned short*)alloc((size_t)MTOT * C * 2);
    float*          actF   = (float*)         alloc((size_t)MTOT * C * 4);
    unsigned short* actBf0 = (unsigned short*)alloc((size_t)MTOT * C * 2);
    unsigned short* actBf1 = (unsigned short*)alloc((size_t)MTOT * C * 2);
    float*          logits = (float*)         alloc((size_t)MTOT * 4);
    float*          vals   = (float*)         alloc(NHEAD * 4);
    int*            sel    = (int*)           alloc(NHEAD * 4);
    unsigned short* gBf    = (unsigned short*)alloc((size_t)NHEAD * C * 2);
    float*          hF     = (float*)         alloc((size_t)NHEAD * C * 4);
    unsigned short* hBf0   = (unsigned short*)alloc((size_t)NHEAD * C * 2);
    unsigned short* hBf1   = (unsigned short*)alloc((size_t)NHEAD * C * 2);
    float*          clsLog = (float*)         alloc((size_t)NHEAD * NCLS * 4);
    float*          boxRaw = (float*)         alloc((size_t)NHEAD * 4 * 4);
    float*          clsArg = (float*)         alloc(NHEAD * 4);

    dim3 w32(32);
    dim3 tblk(32, 8);

    // ---- weight conversion (lat weights already in [N][K] layout; heads transposed) ----
    k_cvt<<<(65536  + 255) / 256, 256, 0, stream>>>(lat_w3, latW3b, 65536);
    k_cvt<<<(131072 + 255) / 256, 256, 0, stream>>>(lat_w4, latW4b, 131072);
    k_cvt<<<(262144 + 255) / 256, 256, 0, stream>>>(lat_w5, latW5b, 262144);
    for (int i = 0; i < 4; ++i) {
        k_cvt_t<<<256, 256, 0, stream>>>(loc_Ws + i * 65536, locWt + i * 65536, 256, 256);
        k_cvt_t<<<256, 256, 0, stream>>>(cls_Ws + i * 65536, clsWt + i * 65536, 256, 256);
        k_cvt_t<<<256, 256, 0, stream>>>(box_Ws + i * 65536, boxWt + i * 65536, 256, 256);
    }
    k_cvt_t<<<1, 256, 0, stream>>>(loc_Wf, locWfT, 256, 1);
    k_cvt_t<<<(256 * NCLS + 255) / 256, 256, 0, stream>>>(cls_Wf, clsWfT, 256, NCLS);
    k_cvt_t<<<4, 256, 0, stream>>>(box_Wf, boxWfT, 256, 4);

    // ---- per-level: transpose to token-major bf16, then WMMA lateral conv + BN ----
    k_transpose<256, 4096><<<dim3(8, 128, BATCH), tblk, 0, stream>>>(x3, xT);
    k_lateral<256, 4096><<<dim3(2048, 4), w32, 0, stream>>>(xT, latW3b, bn_g + 0,   bn_b + 0,
                                                            bn_m + 0,   bn_v + 0,   flatBf, 0);
    k_transpose<512, 1024><<<dim3(16, 32, BATCH), tblk, 0, stream>>>(x4, xT);
    k_lateral<512, 1024><<<dim3(512,  4), w32, 0, stream>>>(xT, latW4b, bn_g + 256, bn_b + 256,
                                                            bn_m + 256, bn_v + 256, flatBf, 4096);
    k_transpose<1024, 256><<<dim3(32, 8, BATCH), tblk, 0, stream>>>(x5, xT);
    k_lateral<1024, 256><<<dim3(128,  4), w32, 0, stream>>>(xT, latW5b, bn_g + 512, bn_b + 512,
                                                            bn_m + 512, bn_v + 512, flatBf, 5120);

    // ---- loc MLP over all 43008 tokens ----
    const unsigned short* Xc = flatBf;
    for (int i = 0; i < 4; ++i) {
        k_linear<4, 256><<<dim3(MTOT / 16, 4), w32, 0, stream>>>(Xc, locWt + i * 65536,
                                                                 loc_bs + i * 256, actF, MTOT, 256);
        unsigned short* dst = (i & 1) ? actBf1 : actBf0;
        k_ln_silu<<<MTOT / 8, 256, 0, stream>>>(actF, loc_lng + i * 256, loc_lnb + i * 256, dst, MTOT);
        Xc = dst;
    }
    k_dot1<<<MTOT / 8, 256, 0, stream>>>(Xc, locWfT, loc_bf, logits, MTOT);

    // ---- top-100 per batch + gather ----
    k_topk<<<BATCH, 256, 0, stream>>>(logits, vals, sel);
    k_gather<<<NHEAD, 256, 0, stream>>>(flatBf, sel, gBf);

    // ---- cls head ----
    Xc = gBf;
    for (int i = 0; i < 4; ++i) {
        k_linear<4, 256><<<dim3(NHEAD / 16, 4), w32, 0, stream>>>(Xc, clsWt + i * 65536,
                                                                  cls_bs + i * 256, hF, NHEAD, 256);
        unsigned short* dst = (i & 1) ? hBf1 : hBf0;
        k_ln_silu<<<NHEAD / 8, 256, 0, stream>>>(hF, cls_lng + i * 256, cls_lnb + i * 256, dst, NHEAD);
        Xc = dst;
    }
    k_linear<1, 256><<<dim3(NHEAD / 16, 5), w32, 0, stream>>>(Xc, clsWfT, cls_bf, clsLog, NHEAD, NCLS);
    k_argmax80<<<NHEAD / 8, 256, 0, stream>>>(clsLog, clsArg, NHEAD);

    // ---- box head ----
    Xc = gBf;
    for (int i = 0; i < 4; ++i) {
        k_linear<4, 256><<<dim3(NHEAD / 16, 4), w32, 0, stream>>>(Xc, boxWt + i * 65536,
                                                                  box_bs + i * 256, hF, NHEAD, 256);
        unsigned short* dst = (i & 1) ? hBf1 : hBf0;
        k_ln_silu<<<NHEAD / 8, 256, 0, stream>>>(hF, box_lng + i * 256, box_lnb + i * 256, dst, NHEAD);
        Xc = dst;
    }
    k_dot4<<<NHEAD / 8, 256, 0, stream>>>(Xc, boxWfT, box_bf, boxRaw, NHEAD);

    // ---- finalize outputs ----
    k_finalize<<<BATCH, 128, 0, stream>>>(vals, sel, clsArg, boxRaw, full_h, full_w, out);
}
